// MaxCutClusters_64828236366589
// MI455X (gfx1250) — matrix operand order repaired and checked by reading, not compiled, not run on hardware
//
#include <hip/hip_runtime.h>
#include <hip/hip_bf16.h>

typedef unsigned short u16;
typedef unsigned int   u32;

typedef __attribute__((ext_vector_type(16))) __bf16 v16bf;
typedef __attribute__((ext_vector_type(8)))  float  v8f;

union FragU { uint4 q[2]; v16bf v; };

__device__ __forceinline__ u16 f2bf(float f) {
    u32 u = __float_as_uint(f);
    return (u16)((u + 0x7fffu + ((u >> 16) & 1u)) >> 16);   // RNE fp32->bf16
}
__device__ __forceinline__ u32 pk2(float a, float b) {
    return (u32)f2bf(a) | ((u32)f2bf(b) << 16);
}
__device__ __forceinline__ uint4 ld8(const float* p) {      // 8 fp32 -> 8 bf16 packed
    const float4* q = (const float4*)p;
    float4 a = q[0], b = q[1];
    uint4 r; r.x = pk2(a.x, a.y); r.y = pk2(a.z, a.w);
    r.z = pk2(b.x, b.y); r.w = pk2(b.z, b.w);
    return r;
}
__device__ __forceinline__ uint4 ld8(const u16* p) {        // 8 bf16 passthrough
    return *(const uint4*)p;
}

__device__ __forceinline__ v8f zero8() {
    v8f z;
#pragma unroll
    for (int i = 0; i < 8; ++i) z[i] = 0.0f;
    return z;
}

// ---- register-staged tile fetch / LDS store -------------------------------
// Tile is RT x 32 bf16 elements; each participating thread owns 8-element
// (16 B) chunks. CH = chunks per thread (compile-time max).
template <int RT, int CH, typename SrcT>
__device__ __forceinline__ void fetch_tile(uint4* reg, const SrcT* src, int ld,
                                           long row0, int k0, long maxRow, int srcK) {
    constexpr int TOTAL = RT * 32;
#pragma unroll
    for (int i = 0; i < CH; ++i) {
        int base = (int)threadIdx.x * 8 + i * 2048;
        if ((TOTAL % 2048 == 0) || base < TOTAL) {
            int r = base >> 5;
            int c = base & 31;
            long gr = row0 + r;
            int  gc = k0 + c;
            uint4 v;
            if (gr < maxRow && gc < srcK) v = ld8(src + gr * (long)ld + gc);
            else { v.x = 0u; v.y = 0u; v.z = 0u; v.w = 0u; }
            reg[i] = v;
        }
    }
}

template <int RT, int CH>
__device__ __forceinline__ void store_tile(u16* lds, const uint4* reg) {
    constexpr int TOTAL = RT * 32;
#pragma unroll
    for (int i = 0; i < CH; ++i) {
        int base = (int)threadIdx.x * 8 + i * 2048;
        if ((TOTAL % 2048 == 0) || base < TOTAL) {
            int r = base >> 5;
            int c = base & 31;
            *(uint4*)(lds + r * 40 + c) = reg[i];
        }
    }
}

// C = leaky( A[M,K] * W[Nout,K]^T + bias ), bf16 WMMA, fp32 accumulate.
// Block: 256 threads (8 waves). Wave w: 16-row strip, BN/16 column subtiles.
// Register-staged double buffering: global loads of tile k+1 overlap WMMAs of
// tile k; staged regs consumed by ds_store only after the next barrier.
template <int BM, int BN, bool RELU, bool OUTBF, typename AT>
__global__ __launch_bounds__(256) void gemm_bf16_wmma(
    const AT* __restrict__ A, int ldA, int KA,
    const float* __restrict__ W, int ldW, int KW, int Nout,
    const float* __restrict__ bias,
    u16* __restrict__ outB, float* __restrict__ outF, int ldc,
    long M, int Kpad, float slope)
{
    constexpr int NT  = BN / 16;
    constexpr int ACH = (BM * 32 + 2047) / 2048;   // 128 -> 2
    constexpr int WCH = (BN * 32 + 2047) / 2048;   // 64 -> 1, 16 -> 1 (partial)

    __shared__ u16 As[BM * 40];
    __shared__ u16 Ws[BN * 40];

    const int tid  = threadIdx.x;
    const int wave = tid >> 5;
    const int lane = tid & 31;
    const int half = lane >> 4;
    const int l15  = lane & 15;
    const long row0 = (long)blockIdx.x * BM;
    const int  n0   = (int)blockIdx.y * BN;

    v8f acc[NT];
#pragma unroll
    for (int i = 0; i < NT; ++i) acc[i] = zero8();

    uint4 aReg[ACH], wReg[WCH];
    fetch_tile<BM, ACH>(aReg, A, ldA, row0, 0, M, KA);
    fetch_tile<BN, WCH>(wReg, W, ldW, (long)n0, 0, (long)Nout, KW);

    for (int k0 = 0; k0 < Kpad; k0 += 32) {
        store_tile<BM, ACH>(As, aReg);
        store_tile<BN, WCH>(Ws, wReg);
        __syncthreads();

        // Issue next tile's global loads; results consumed after next barrier.
        if (k0 + 32 < Kpad) {
            fetch_tile<BM, ACH>(aReg, A, ldA, row0, k0 + 32, M, KA);
            fetch_tile<BN, WCH>(wReg, W, ldW, (long)n0, k0 + 32, (long)Nout, KW);
            if (k0 + 64 < Kpad) {  // prefetch A stream 2 tiles ahead (global_prefetch_b8)
                long gr = row0 + (((int)tid * 8) >> 5);
                if (gr < M)
                    __builtin_prefetch(A + gr * (long)ldA + (k0 + 64), 0, 1);
            }
        }

        // Batch all LDS fragment loads, then run WMMAs back-to-back.
        // A fragment: lane<16 -> M=l15, K {8h..8h+7} then {16+8h..23+8h}
        FragU af;
        const u16* ap = &As[(wave * 16 + l15) * 40 + 8 * half];
        af.q[0] = *(const uint4*)(ap);
        af.q[1] = *(const uint4*)(ap + 16);
        FragU bf[NT];
#pragma unroll
        for (int nt = 0; nt < NT; ++nt) {
            // B fragment: N = l15, K {8h..8h+7} then {16+8h..23+8h}
            const u16* bp = &Ws[(nt * 16 + l15) * 40 + 8 * half];
            bf[nt].q[0] = *(const uint4*)(bp);
            bf[nt].q[1] = *(const uint4*)(bp + 16);
        }
#pragma unroll
        for (int nt = 0; nt < NT; ++nt) {
            acc[nt] = __builtin_amdgcn_wmma_f32_16x16x32_bf16(
                false, af.v, false, bf[nt].v, (short)0, acc[nt], false, false);
        }
        __syncthreads();
    }

    // Epilogue: C/D layout -> VGPR r, lane: M = r + 8*half, N = l15
#pragma unroll
    for (int nt = 0; nt < NT; ++nt) {
        const int col = n0 + nt * 16 + l15;
        const float bv = bias[col];
#pragma unroll
        for (int r = 0; r < 8; ++r) {
            long m = row0 + wave * 16 + r + 8 * half;
            if (m < M) {
                float v = acc[nt][r] + bv;
                if (RELU) v = (v >= 0.0f) ? v : v * slope;
                if (OUTBF) outB[m * (long)ldc + col] = f2bf(v);
                else       outF[m * (long)ldc + col] = v;
            }
        }
    }
}

// Q = softmax(clusters) per row of 16; write fp32 Q and bf16 Q into concat buffer.
__global__ void softmax_q_kernel(const float* __restrict__ clusters,
                                 float* __restrict__ Qf, u16* __restrict__ zbf,
                                 long N, int ldz, int hofs) {
    long n = (long)blockIdx.x * 256 + threadIdx.x;
    if (n >= N) return;
    const float4* p = (const float4*)(clusters + n * 16);
    float f[16];
    float4 a = p[0], b = p[1], c = p[2], d = p[3];
    f[0]=a.x; f[1]=a.y; f[2]=a.z; f[3]=a.w; f[4]=b.x; f[5]=b.y; f[6]=b.z; f[7]=b.w;
    f[8]=c.x; f[9]=c.y; f[10]=c.z; f[11]=c.w; f[12]=d.x; f[13]=d.y; f[14]=d.z; f[15]=d.w;
    float m = f[0];
#pragma unroll
    for (int j = 1; j < 16; ++j) m = fmaxf(m, f[j]);
    float s = 0.0f;
#pragma unroll
    for (int j = 0; j < 16; ++j) { f[j] = __expf(f[j] - m); s += f[j]; }
    float inv = 1.0f / s;
#pragma unroll
    for (int j = 0; j < 16; ++j) f[j] *= inv;
    float4* q = (float4*)(Qf + n * 16);
    q[0] = make_float4(f[0], f[1], f[2], f[3]);
    q[1] = make_float4(f[4], f[5], f[6], f[7]);
    q[2] = make_float4(f[8], f[9], f[10], f[11]);
    q[3] = make_float4(f[12], f[13], f[14], f[15]);
    uint4* z = (uint4*)(zbf + n * (long)ldz + hofs);
    uint4 z0, z1;
    z0.x = pk2(f[0], f[1]);  z0.y = pk2(f[2], f[3]);
    z0.z = pk2(f[4], f[5]);  z0.w = pk2(f[6], f[7]);
    z1.x = pk2(f[8], f[9]);  z1.y = pk2(f[10], f[11]);
    z1.z = pk2(f[12], f[13]); z1.w = pk2(f[14], f[15]);
    z[0] = z0; z[1] = z1;
}

__global__ void deg_count_kernel(const int* __restrict__ col, float* __restrict__ deg, long E) {
    long e = (long)blockIdx.x * 256 + threadIdx.x;
    if (e >= E) return;
    atomicAdd(&deg[col[e]], 1.0f);
}

__global__ void dinv_kernel(const float* __restrict__ deg, float* __restrict__ dinv, long N) {
    long n = (long)blockIdx.x * 256 + threadIdx.x;
    if (n >= N) return;
    dinv[n] = rsqrtf(deg[n] + 1.0f);   // +1 self loop
}

// Per edge: w = dinv[r]*dinv[c]; deg_lap[r]+=w; AQ[r,:]+=w*Q[c,:]
// (self loops cancel exactly in Q^T L Q, so they are skipped)
__global__ void lap_edge_kernel(const int* __restrict__ row, const int* __restrict__ col,
                                const float* __restrict__ dinv, const float* __restrict__ Qf,
                                float* __restrict__ deg_lap, float* __restrict__ AQ, long E) {
    long e = (long)blockIdx.x * 256 + threadIdx.x;
    if (e >= E) return;
    int r = row[e], c = col[e];
    float w = dinv[r] * dinv[c];
    atomicAdd(&deg_lap[r], w);
    const float4* q = (const float4*)(Qf + (long)c * 16);
    float* aq = AQ + (long)r * 16;
    float4 q0 = q[0], q1 = q[1], q2 = q[2], q3 = q[3];
    atomicAdd(aq + 0,  w * q0.x); atomicAdd(aq + 1,  w * q0.y);
    atomicAdd(aq + 2,  w * q0.z); atomicAdd(aq + 3,  w * q0.w);
    atomicAdd(aq + 4,  w * q1.x); atomicAdd(aq + 5,  w * q1.y);
    atomicAdd(aq + 6,  w * q1.z); atomicAdd(aq + 7,  w * q1.w);
    atomicAdd(aq + 8,  w * q2.x); atomicAdd(aq + 9,  w * q2.y);
    atomicAdd(aq + 10, w * q2.z); atomicAdd(aq + 11, w * q2.w);
    atomicAdd(aq + 12, w * q3.x); atomicAdd(aq + 13, w * q3.y);
    atomicAdd(aq + 14, w * q3.z); atomicAdd(aq + 15, w * q3.w);
}

// loss = sum_n sum_j Q[n,j] * (deg_lap[n]*Q[n,j] - AQ[n,j])
__global__ void loss_kernel(const float* __restrict__ Qf, const float* __restrict__ deg_lap,
                            const float* __restrict__ AQ, float* __restrict__ loss, long N) {
    __shared__ float red[256];
    long n = (long)blockIdx.x * 256 + threadIdx.x;
    float acc = 0.0f;
    if (n < N) {
        float dl = deg_lap[n];
        const float4* q  = (const float4*)(Qf + n * 16);
        const float4* aq = (const float4*)(AQ + n * 16);
#pragma unroll
        for (int i = 0; i < 4; ++i) {
            float4 qq = q[i], a = aq[i];
            acc += qq.x * (dl * qq.x - a.x) + qq.y * (dl * qq.y - a.y)
                 + qq.z * (dl * qq.z - a.z) + qq.w * (dl * qq.w - a.w);
        }
    }
    red[threadIdx.x] = acc;
    __syncthreads();
    for (int s = 128; s > 0; s >>= 1) {
        if ((int)threadIdx.x < s) red[threadIdx.x] += red[threadIdx.x + s];
        __syncthreads();
    }
    if (threadIdx.x == 0) atomicAdd(loss, red[0]);
}

__global__ void logsoftmax_kernel(float* __restrict__ out, long N) {
    long n = (long)blockIdx.x * 256 + threadIdx.x;
    if (n >= N) return;
    float4* p = (float4*)(out + n * 16);
    float f[16];
    float4 a = p[0], b = p[1], c = p[2], d = p[3];
    f[0]=a.x; f[1]=a.y; f[2]=a.z; f[3]=a.w; f[4]=b.x; f[5]=b.y; f[6]=b.z; f[7]=b.w;
    f[8]=c.x; f[9]=c.y; f[10]=c.z; f[11]=c.w; f[12]=d.x; f[13]=d.y; f[14]=d.z; f[15]=d.w;
    float m = f[0];
#pragma unroll
    for (int j = 1; j < 16; ++j) m = fmaxf(m, f[j]);
    float s = 0.0f;
#pragma unroll
    for (int j = 0; j < 16; ++j) s += __expf(f[j] - m);
    float l = m + __logf(s);
#pragma unroll
    for (int j = 0; j < 16; ++j) f[j] -= l;
    p[0] = make_float4(f[0], f[1], f[2], f[3]);
    p[1] = make_float4(f[4], f[5], f[6], f[7]);
    p[2] = make_float4(f[8], f[9], f[10], f[11]);
    p[3] = make_float4(f[12], f[13], f[14], f[15]);
}

extern "C" void kernel_launch(void* const* d_in, const int* in_sizes, int n_in,
                              void* d_out, int out_size, void* d_ws, size_t ws_size,
                              hipStream_t stream) {
    // Inputs: 0 x, 1 edge_index, 2 W_enc, 3 b_enc, 4 clusters, 5 W1, 6 b1,
    //         7 W2, 8 b2, 9 W_dec, 10 b_dec
    const float* x        = (const float*)d_in[0];
    const int*   ei       = (const int*)d_in[1];
    const float* W_enc    = (const float*)d_in[2];
    const float* b_enc    = (const float*)d_in[3];
    const float* clusters = (const float*)d_in[4];
    const float* W1       = (const float*)d_in[5];
    const float* b1       = (const float*)d_in[6];
    const float* W2       = (const float*)d_in[7];
    const float* b2       = (const float*)d_in[8];
    const float* W_dec    = (const float*)d_in[9];
    const float* b_dec    = (const float*)d_in[10];

    const int  H   = in_sizes[3];                 // 256
    const int  FIN = in_sizes[2] / H;             // 512
    const long N   = (long)in_sizes[0] / FIN;     // 100000
    const long E   = (long)in_sizes[1] / 2;       // 3.2M
    const int  C   = in_sizes[10];                // 16
    const int  H4  = in_sizes[6];                 // 1024
    const int  KZ  = H + C;                       // 272
    const int  KZp = ((KZ + 31) / 32) * 32;       // 288

    // Workspace carve-up (256B aligned)
    size_t off = 0;
    auto carve = [&](size_t bytes) {
        size_t r = off;
        off = (off + bytes + 255) & ~(size_t)255;
        return r;
    };
    char* ws = (char*)d_ws;
    float* Qf      = (float*)(ws + carve((size_t)N * C * 4));
    float* deg     = (float*)(ws + carve((size_t)N * 4));
    float* dinv    = (float*)(ws + carve((size_t)N * 4));
    float* deg_lap = (float*)(ws + carve((size_t)N * 4));
    float* AQ      = (float*)(ws + carve((size_t)N * C * 4));
    u16*   zbf     = (u16*)  (ws + carve((size_t)N * KZ * 2));   // [N, H+C] bf16
    u16*   z1bf    = (u16*)  (ws + carve((size_t)N * H4 * 2));   // [N, 4H]  bf16
    u16*   z2bf    = (u16*)  (ws + carve((size_t)N * H  * 2));   // [N, H]   bf16
    (void)ws_size; (void)n_in; (void)out_size;

    float* outf = (float*)d_out;                  // [N,C] logits -> log_softmax
    float* loss = outf + (size_t)N * C;           // scalar

    // Reset accumulators (graph-capturable async memsets)
    hipMemsetAsync(deg,     0, (size_t)N * 4,     stream);
    hipMemsetAsync(deg_lap, 0, (size_t)N * 4,     stream);
    hipMemsetAsync(AQ,      0, (size_t)N * C * 4, stream);
    hipMemsetAsync(loss,    0, sizeof(float),     stream);

    const dim3 blk(256);
    const int gN = (int)((N + 255) / 256);
    const int gE = (int)((E + 255) / 256);
    const int gM = (int)((N + 127) / 128);

    // Graph path
    softmax_q_kernel<<<gN, blk, 0, stream>>>(clusters, Qf, zbf, N, KZ, H);
    deg_count_kernel<<<gE, blk, 0, stream>>>(ei + E, deg, E);
    dinv_kernel<<<gN, blk, 0, stream>>>(deg, dinv, N);
    lap_edge_kernel<<<gE, blk, 0, stream>>>(ei, ei + E, dinv, Qf, deg_lap, AQ, E);
    loss_kernel<<<gN, blk, 0, stream>>>(Qf, deg_lap, AQ, loss, N);

    // GEMM chain (bf16 WMMA, fp32 accumulate)
    // h = x @ W_enc^T + b_enc -> zbf cols [0, H)
    gemm_bf16_wmma<128, 64, false, true, float><<<dim3(gM, H / 64), blk, 0, stream>>>(
        x, FIN, FIN, W_enc, FIN, FIN, H, b_enc, zbf, nullptr, KZ, N, FIN, 0.0f);
    // z1 = leaky(z @ W1^T + b1), K padded 272 -> 288
    gemm_bf16_wmma<128, 64, true, true, u16><<<dim3(gM, H4 / 64), blk, 0, stream>>>(
        zbf, KZ, KZ, W1, KZ, KZ, H4, b1, z1bf, nullptr, H4, N, KZp, (float)H4);
    // z2 = z1 @ W2^T + b2
    gemm_bf16_wmma<128, 64, false, true, u16><<<dim3(gM, H / 64), blk, 0, stream>>>(
        z1bf, H4, H4, W2, H4, H4, H, b2, z2bf, nullptr, H, N, H4, 0.0f);
    // logits = z2 @ W_dec^T + b_dec (fp32 out, BN=16)
    gemm_bf16_wmma<128, 16, false, false, u16><<<dim3(gM, C / 16), blk, 0, stream>>>(
        z2bf, H, H, W_dec, H, H, C, b_dec, nullptr, outf, C, N, H, 0.0f);

    logsoftmax_kernel<<<gN, blk, 0, stream>>>(outf, N);
}
